// NoisyKGate_9268539425526
// MI455X (gfx1250) — matrix operand order repaired and checked by reading, not compiled, not run on hardware
//
#include <hip/hip_runtime.h>
#include <hip/hip_bf16.h>
#include <math.h>

// MoE router gate: s = sigmoid(x @ W + b); top-8 of 64 experts; normalize gates.
// B=4, S=4096, D=4096, E=64, K=8.  Memory-bound (268 MB of x @ 23.3 TB/s ~= 12us),
// needs ~730 TFLOP/s matrix rate -> bf16 WMMA (V_WMMA_F32_16X16X32_BF16), fp32 acc.
// Inner loop: explicit 2-stage ping-pong (K-step 64) so each stage's global loads
// land directly in the registers the *next* stage's WMMAs consume -- no v_mov
// shuffles, and loads get a full stage of latency before their s_wait.

typedef __attribute__((ext_vector_type(16))) __bf16 v16bf;
typedef __attribute__((ext_vector_type(8)))  float  v8f;

constexpr int Bc = 4, Sc = 4096, Dc = 4096, Ec = 64, Kc = 8;
constexpr int Tc = Bc * Sc;              // 16384 tokens

// ---------------------------------------------------------------------------
// Kernel 1: W [D,E] fp32 row-major  ->  Wt [E,D] bf16 (transposed, contiguous K)
// so B-fragments are two contiguous b128 loads per lane, no strided gathers.
// ---------------------------------------------------------------------------
__global__ void wcvt_kernel(const float* __restrict__ W, __bf16* __restrict__ Wt) {
    int tid = blockIdx.x * blockDim.x + threadIdx.x;   // 0 .. E*D-1
    int n = tid >> 12;                                 // expert (0..63)
    int k = tid & (Dc - 1);                            // depth  (0..4095)
    Wt[(size_t)n * Dc + k] = (__bf16)W[(size_t)k * Ec + n];
}

// ---------------------------------------------------------------------------
// Kernel 2: GEMM + sigmoid.  Wave = 16 tokens x 64 experts.
// A fragment (16x32 bf16, ISA layout): lane l (lo half) holds row M=l,
//   halves 0-7 = K..K+7, halves 8-15 = K+16..K+23; hi half lanes hold K+8.. / K+24..
// B fragment: lane = column N of Wt, 8 bf16 at K + 8*(lane>=16) and 8 at K + 16 + 8*(lane>=16).
// ---------------------------------------------------------------------------
struct KTile {
    float4 f0, f1, f2, f3;   // A: 16 fp32 for this lane
    uint4  b[4][2];          // B: 4 expert tiles x 16 bf16
};

__global__ __launch_bounds__(256) void
gate_gemm_kernel(const float* __restrict__ x, const __bf16* __restrict__ Wt,
                 const float* __restrict__ bias, float* __restrict__ s_out) {
    const int lane = threadIdx.x & 31;
    const int wave = threadIdx.x >> 5;
    const int hi   = lane >> 4;          // 0: lanes 0-15, 1: lanes 16-31
    const int l15  = lane & 15;

    const int trow = blockIdx.x * 128 + wave * 16;     // first token of this wave
    const float* __restrict__ xrow = x + (size_t)(trow + l15) * Dc + hi * 8;

    const __bf16* __restrict__ wp[4];
#pragma unroll
    for (int e = 0; e < 4; ++e)
        wp[e] = Wt + (size_t)(e * 16 + l15) * Dc + hi * 8;

    v8f acc[4];
#pragma unroll
    for (int e = 0; e < 4; ++e) acc[e] = (v8f)(0.0f);

    auto load_tile = [&](KTile& t, int kk) {
        const float* pA = xrow + kk;
        t.f0 = *(const float4*)(pA +  0);
        t.f1 = *(const float4*)(pA +  4);
        t.f2 = *(const float4*)(pA + 16);
        t.f3 = *(const float4*)(pA + 20);
#pragma unroll
        for (int e = 0; e < 4; ++e) {
            const __bf16* pB = wp[e] + kk;
            t.b[e][0] = *(const uint4*)(pB + 0);
            t.b[e][1] = *(const uint4*)(pB + 16);
        }
    };

    auto compute_tile = [&](const KTile& t) {
        v16bf a;
        a[0]  = (__bf16)t.f0.x; a[1]  = (__bf16)t.f0.y; a[2]  = (__bf16)t.f0.z; a[3]  = (__bf16)t.f0.w;
        a[4]  = (__bf16)t.f1.x; a[5]  = (__bf16)t.f1.y; a[6]  = (__bf16)t.f1.z; a[7]  = (__bf16)t.f1.w;
        a[8]  = (__bf16)t.f2.x; a[9]  = (__bf16)t.f2.y; a[10] = (__bf16)t.f2.z; a[11] = (__bf16)t.f2.w;
        a[12] = (__bf16)t.f3.x; a[13] = (__bf16)t.f3.y; a[14] = (__bf16)t.f3.z; a[15] = (__bf16)t.f3.w;
#pragma unroll
        for (int e = 0; e < 4; ++e) {
            union { uint4 u[2]; v16bf v; } bb;
            bb.u[0] = t.b[e][0];
            bb.u[1] = t.b[e][1];
            acc[e] = __builtin_amdgcn_wmma_f32_16x16x32_bf16(
                false, a, false, bb.v, (short)0, acc[e], false, false);
        }
    };

    // ---- explicit 2-stage software pipeline, K-step 64 per iteration ----
    KTile t0, t1;
    load_tile(t0, 0);
    for (int k0 = 0; k0 < Dc; k0 += 64) {
        load_tile(t1, k0 + 32);                        // always < Dc
        __builtin_prefetch(xrow + k0 + 512, 0, 3);     // x stream look-ahead
        compute_tile(t0);                              // k0
        const int kn = (k0 + 64 < Dc) ? (k0 + 64) : 0; // wrap: redundant, in-bounds
        load_tile(t0, kn);
        compute_tile(t1);                              // k0 + 32
    }

    // ---- epilogue: bias + sigmoid, store s.  C layout: M = r + 8*hi, N = l15 ----
#pragma unroll
    for (int e = 0; e < 4; ++e) {
        const int   col = e * 16 + l15;
        const float bv  = bias[col];
#pragma unroll
        for (int r = 0; r < 8; ++r) {
            const int tok = trow + 8 * hi + r;
            float z = acc[e][r] + bv;
            float s = 1.0f / (1.0f + __expf(-z));
            s_out[(size_t)tok * Ec + col] = s;
        }
    }
}

// ---------------------------------------------------------------------------
// Kernel 3: per-token top-8 of 64 + gate normalization.
// One thread per token; row held in registers; fully-unrolled static indexing.
// Tie-break matches jax.lax.top_k: value descending, index ascending.
// ---------------------------------------------------------------------------
__global__ __launch_bounds__(256) void
topk_kernel(const float* __restrict__ s, float* __restrict__ g, float* __restrict__ idxf) {
    const int t = blockIdx.x * blockDim.x + threadIdx.x;
    if (t >= Tc) return;

    float v[Ec];
    const float4* p = (const float4*)(s + (size_t)t * Ec);
#pragma unroll
    for (int i = 0; i < Ec / 4; ++i) {
        float4 q = p[i];
        v[4 * i + 0] = q.x; v[4 * i + 1] = q.y; v[4 * i + 2] = q.z; v[4 * i + 3] = q.w;
    }

    float gv[Kc]; int gi[Kc];
    float pv = INFINITY; int pi = -1;    // previous selection (exclusion threshold)
    float sum = 0.0f;
#pragma unroll
    for (int j = 0; j < Kc; ++j) {
        float best = -INFINITY; int bi = 0;
#pragma unroll
        for (int i = 0; i < Ec; ++i) {
            const float vi = v[i];
            const bool eligible = (vi < pv) || (vi == pv && i > pi);
            if (eligible && vi > best) { best = vi; bi = i; }
        }
        gv[j] = best; gi[j] = bi;
        pv = best; pi = bi;
        sum += best;
    }
    const float inv = 1.0f / sum;
#pragma unroll
    for (int j = 0; j < Kc; ++j) {
        g[(size_t)t * Kc + j]    = gv[j] * inv;
        idxf[(size_t)t * Kc + j] = (float)gi[j];   // indices in fp32 output buffer
    }
}

// ---------------------------------------------------------------------------
extern "C" void kernel_launch(void* const* d_in, const int* in_sizes, int n_in,
                              void* d_out, int out_size, void* d_ws, size_t ws_size,
                              hipStream_t stream) {
    const float* x    = (const float*)d_in[0];   // [4,4096,4096]
    const float* W    = (const float*)d_in[1];   // [4096,64]
    const float* bias = (const float*)d_in[2];   // [64]
    // d_in[3] = k (scalar 8), compile-time constant here.

    __bf16* Wt = (__bf16*)d_ws;                  // 4096*64*2 = 512 KB scratch

    float* out  = (float*)d_out;
    float* g    = out;                           // [16384, 8]
    float* idxf = out + (size_t)Tc * Kc;         // [16384, 8]
    float* sbuf = out + (size_t)2 * Tc * Kc;     // [16384, 64]

    // 1) W -> bf16 transposed (L2-resident thereafter)
    hipLaunchKernelGGL(wcvt_kernel, dim3((Dc * Ec) / 256), dim3(256), 0, stream, W, Wt);
    // 2) streaming GEMM + sigmoid: 128 blocks x 8 waves x (16 tok x 64 exp)
    hipLaunchKernelGGL(gate_gemm_kernel, dim3(Tc / 128), dim3(256), 0, stream,
                       x, Wt, bias, sbuf);
    // 3) top-8 + normalize
    hipLaunchKernelGGL(topk_kernel, dim3(Tc / 256), dim3(256), 0, stream, sbuf, g, idxf);
}
